// Mamba3Block_6674379178502
// MI455X (gfx1250) — compile-verified, matrix operand order checked by reading, not run on hardware
//
#include <hip/hip_runtime.h>
#include <hip/hip_bf16.h>

// ---------------------------------------------------------------------------
// Mamba3 block for MI455X (gfx1250, wave32, WMMA).
// Big GEMMs: async-copy tiles to LDS (ASYNCcnt) -> ds_load_b128 fragments ->
// v_wmma_f32_16x16x32_bf16 (f32 accumulate). SSM scan keeps h[16] in VGPRs and
// recomputes dA/u on the fly (never materializes the (B,L,E,S) tensors).
// ---------------------------------------------------------------------------

#define SEQL    2048
#define DMODEL  1024
#define DINNER  2048
#define DSTATE  16
#define NROWS   4096          // B * SEQ
#define A_FLOOR 1e-4f
#define RMS_EPS 1.1920928955078125e-07f

// LDS tile pitch in bf16 elements: 32 K + 8 pad -> 80B row pitch, which maps
// lane fragment rows onto all sixteen 4-bank-aligned starts (conflict-free).
#define LDS_PITCH 40

typedef __attribute__((ext_vector_type(8)))  __bf16 v8bf;
typedef __attribute__((ext_vector_type(16))) __bf16 v16bf;
typedef __attribute__((ext_vector_type(8)))  float  v8f;
typedef __attribute__((ext_vector_type(4)))  int    v4i;
typedef __attribute__((address_space(1)))    v4i    gv4i;   // global int4
typedef __attribute__((address_space(3)))    v4i    lv4i;   // LDS int4

#if defined(__gfx1250__) && __has_builtin(__builtin_amdgcn_global_load_async_to_lds_b128)
#define USE_ASYNC_LDS 1
#else
#define USE_ASYNC_LDS 0
#endif

__device__ __forceinline__ unsigned short f32_to_bf16(float f) {
    union { float f; unsigned int u; } v; v.f = f;
    unsigned int r = v.u + 0x7fffu + ((v.u >> 16) & 1u);   // RNE
    return (unsigned short)(r >> 16);
}
__device__ __forceinline__ float softplusf(float x) {
    return (x > 20.f) ? x : log1pf(expf(x));
}
__device__ __forceinline__ float sigmoidf(float x) {
    return 1.f / (1.f + expf(-x));
}

// 16-byte global -> LDS copy (async on CDNA5 if the builtin is available).
__device__ __forceinline__ void copy16_g2l(const unsigned short* __restrict__ g,
                                           unsigned short* l) {
#if USE_ASYNC_LDS
    __builtin_amdgcn_global_load_async_to_lds_b128(
        (gv4i*)(unsigned short*)g, (lv4i*)l, 0, 0);
#else
    *reinterpret_cast<uint4*>(l) = *reinterpret_cast<const uint4*>(g);
#endif
}
__device__ __forceinline__ void wait_copies_and_barrier() {
#if USE_ASYNC_LDS
#if __has_builtin(__builtin_amdgcn_s_wait_asynccnt)
    __builtin_amdgcn_s_wait_asynccnt(0);
#else
    asm volatile("s_wait_asynccnt 0x0" ::: "memory");
#endif
#endif
    __syncthreads();
}

// --------------------------- fp32 -> bf16 prepass ---------------------------
__global__ __launch_bounds__(256) void convert_bf16_kernel(
        const float* __restrict__ in, unsigned short* __restrict__ out, int n) {
    int i = blockIdx.x * blockDim.x + threadIdx.x;
    int stride = gridDim.x * blockDim.x;
    for (; i < n; i += stride) out[i] = f32_to_bf16(in[i]);
}

// ------------------------------ WMMA GEMM -----------------------------------
// C[M,N] = A[M,K] * B[N,K]^T, row-major bf16 operands, f32 accum.
// Fragment lane L: row = base + (L&15); K runs [8*(L>>4),+8) and [16+8*(L>>4),+8).
__device__ __forceinline__ v16bf frag_from_lds(const unsigned short* p) {
    v8bf lo = *reinterpret_cast<const v8bf*>(p);
    v8bf hi = *reinterpret_cast<const v8bf*>(p + 16);
    return __builtin_shufflevector(lo, hi, 0,1,2,3,4,5,6,7,8,9,10,11,12,13,14,15);
}

// MODE 0: GEMM1 epilogue — cols [0,2048) -> silu -> x_path ; cols [2048,4096) -> z (raw)
// MODE 1: GEMM2 epilogue — plain f32 store to out0 with ld = N
template <int MODE>
__global__ __launch_bounds__(256) void wmma_gemm_kernel(
        const unsigned short* __restrict__ A, const unsigned short* __restrict__ B,
        float* __restrict__ out0, float* __restrict__ out1,
        int M, int N, int K) {
    __shared__ unsigned short ldsA[128 * LDS_PITCH];
    __shared__ unsigned short ldsB[64 * LDS_PITCH];

    const int tid   = threadIdx.x;
    const int lane  = tid & 31;
    const int wave  = tid >> 5;                  // 8 waves, stacked along M
    const int fr    = lane & 15;                 // fragment row/col within tile
    const int khalf = lane >> 4;
    const int mbase = blockIdx.x * 128;
    const int n0    = blockIdx.y * 64;

    // cooperative tile copy assignment: 16B chunks (row, 8-elem k offset)
    const int crow = tid >> 2;                   // 0..63
    const int cko  = (tid & 3) * 8;              // 0,8,16,24

    v8f acc[4] = {};
    for (int kb = 0; kb < K; kb += 32) {
        // A tile: 128 rows x 32 k  (2 chunks/thread), B tile: 64 rows x 32 k
        copy16_g2l(A + (size_t)(mbase + crow) * K + kb + cko,
                   &ldsA[crow * LDS_PITCH + cko]);
        copy16_g2l(A + (size_t)(mbase + crow + 64) * K + kb + cko,
                   &ldsA[(crow + 64) * LDS_PITCH + cko]);
        copy16_g2l(B + (size_t)(n0 + crow) * K + kb + cko,
                   &ldsB[crow * LDS_PITCH + cko]);
        wait_copies_and_barrier();

        v16bf afrag = frag_from_lds(&ldsA[(wave * 16 + fr) * LDS_PITCH + 8 * khalf]);
#pragma unroll
        for (int t = 0; t < 4; ++t) {
            v16bf bfrag = frag_from_lds(&ldsB[(t * 16 + fr) * LDS_PITCH + 8 * khalf]);
            acc[t] = __builtin_amdgcn_wmma_f32_16x16x32_bf16(
                false, afrag, false, bfrag, (short)0, acc[t], false, false);
        }
        __syncthreads();
    }

    // D layout: lane holds column (lane&15) of rows 8*(lane>>4)+r, r=0..7
    const int rbase = mbase + wave * 16 + khalf * 8;
#pragma unroll
    for (int t = 0; t < 4; ++t) {
        const int col = n0 + t * 16 + fr;
#pragma unroll
        for (int r = 0; r < 8; ++r) {
            const int row = rbase + r;
            float v = acc[t][r];
            if (MODE == 0) {
                if (col < DINNER) out0[(size_t)row * DINNER + col] = v * sigmoidf(v);
                else              out1[(size_t)row * DINNER + (col - DINNER)] = v;
            } else {
                out0[(size_t)row * N + col] = v;
            }
        }
    }
}

// ------------------- per-row ssm projection + norms + dt ---------------------
// One block (256 threads) per (b,l) row: ssm[34] = x_path_row · W_x^T, then
// rmsnorm(B+bias), rmsnorm(C+bias), dt_in, lam=sigmoid, mean_e softplus(dt).
__global__ __launch_bounds__(256) void ssm_row_kernel(
        const float* __restrict__ xpath, const float* __restrict__ Wx,
        const float* __restrict__ W_dt, const float* __restrict__ b_dt,
        const float* __restrict__ B_bias, const float* __restrict__ C_bias,
        const float* __restrict__ B_norm_w, const float* __restrict__ C_norm_w,
        float* __restrict__ Bn, float* __restrict__ Cn,
        float* __restrict__ dt_in_o, float* __restrict__ lam_o,
        float* __restrict__ mean_dt_o) {
    const int row = blockIdx.x;
    const int t   = threadIdx.x;
    __shared__ float acc[36];
    __shared__ float msum;
    if (t < 36) acc[t] = 0.f;
    if (t == 0) msum = 0.f;
    __syncthreads();

    float xr[8];
    const float* xp = xpath + (size_t)row * DINNER + t * 8;
#pragma unroll
    for (int j = 0; j < 8; ++j) xr[j] = xp[j];

    for (int s = 0; s < 34; ++s) {
        const float* w = Wx + (size_t)s * DINNER + t * 8;
        float p = 0.f;
#pragma unroll
        for (int j = 0; j < 8; ++j) p += xr[j] * w[j];
#pragma unroll
        for (int off = 16; off; off >>= 1) p += __shfl_down(p, off, 32);
        if ((t & 31) == 0) atomicAdd(&acc[s], p);
    }
    __syncthreads();

    const float dti = acc[32];
    {   // mean over e of softplus(dt_in * W_dt[e] + b_dt[e])
        float p = 0.f;
#pragma unroll
        for (int j = 0; j < 8; ++j) {
            int e = t * 8 + j;
            p += softplusf(dti * W_dt[e] + b_dt[e]);
        }
#pragma unroll
        for (int off = 16; off; off >>= 1) p += __shfl_down(p, off, 32);
        if ((t & 31) == 0) atomicAdd(&msum, p);
    }
    __syncthreads();

    if (t < 16) {
        float ss = 0.f;
#pragma unroll
        for (int s = 0; s < 16; ++s) { float v = acc[s] + B_bias[s]; ss += v * v; }
        float v = acc[t] + B_bias[t];
        Bn[(size_t)row * 16 + t] = v * rsqrtf(ss * (1.f / 16.f) + RMS_EPS) * B_norm_w[t];
    } else if (t < 32) {
        int i = t - 16;
        float ss = 0.f;
#pragma unroll
        for (int s = 0; s < 16; ++s) { float v = acc[16 + s] + C_bias[s]; ss += v * v; }
        float v = acc[16 + i] + C_bias[i];
        Cn[(size_t)row * 16 + i] = v * rsqrtf(ss * (1.f / 16.f) + RMS_EPS) * C_norm_w[i];
    } else if (t == 32) {
        dt_in_o[row] = dti;
    } else if (t == 33) {
        lam_o[row] = sigmoidf(acc[33]);
    } else if (t == 34) {
        mean_dt_o[row] = msum * (1.f / (float)DINNER);
    }
}

// ------------------------- cumsum over sequence -----------------------------
__global__ void cumsum_kernel(const float* __restrict__ mean_dt,
                              float* __restrict__ cum_dt) {
    int b = threadIdx.x;
    if (b < 2) {
        float run = 0.f;
        for (int l = 0; l < SEQL; ++l) {
            run += mean_dt[b * SEQL + l];
            cum_dt[b * SEQL + l] = run;
        }
    }
}

// ------------------------------- RoPE ---------------------------------------
__global__ __launch_bounds__(256) void rope_kernel(
        const float* __restrict__ Bn, const float* __restrict__ Cn,
        const float* __restrict__ cum_dt, const float* __restrict__ rope_freqs,
        float* __restrict__ Br, float* __restrict__ Cr) {
    int row = blockIdx.x * blockDim.x + threadIdx.x;
    if (row >= NROWS) return;
    float cd = cum_dt[row];
#pragma unroll
    for (int i = 0; i < 8; ++i) {
        float f   = softplusf(rope_freqs[i]);
        float ang = cd * f;
        float c = cosf(ang), s = sinf(ang);
        float be = Bn[(size_t)row * 16 + 2 * i], bo = Bn[(size_t)row * 16 + 2 * i + 1];
        Br[(size_t)row * 16 + 2 * i]     = be * c - bo * s;
        Br[(size_t)row * 16 + 2 * i + 1] = be * s + bo * c;
        float ce = Cn[(size_t)row * 16 + 2 * i], co = Cn[(size_t)row * 16 + 2 * i + 1];
        Cr[(size_t)row * 16 + 2 * i]     = ce * c - co * s;
        Cr[(size_t)row * 16 + 2 * i + 1] = ce * s + co * c;
    }
}

// ----------------------------- SSM scan -------------------------------------
// One thread per (b,e); h[16], Bx_prev[16], A[16] live in VGPRs.
// dA and u are recomputed per step. Fuses D-skip + silu(z) gate, emits bf16.
__global__ __launch_bounds__(256) void scan_kernel(
        const float* __restrict__ xpath, const float* __restrict__ zbuf,
        const float* __restrict__ Br, const float* __restrict__ Cr,
        const float* __restrict__ dt_in, const float* __restrict__ lam_arr,
        const float* __restrict__ W_dt, const float* __restrict__ b_dt,
        const float* __restrict__ A_log, const float* __restrict__ Dvec,
        unsigned short* __restrict__ y_gated) {
    const int b = blockIdx.x >> 3;
    const int e = (blockIdx.x & 7) * 256 + threadIdx.x;

    float A[DSTATE];
#pragma unroll
    for (int s = 0; s < DSTATE; ++s)
        A[s] = fminf(-expf(A_log[(size_t)e * DSTATE + s]), -A_FLOOR);

    const float wdt = W_dt[e], bdt = b_dt[e], Dv = Dvec[e];
    float h[DSTATE], bxp[DSTATE];
#pragma unroll
    for (int s = 0; s < DSTATE; ++s) { h[s] = 0.f; bxp[s] = 0.f; }

    for (int l = 0; l < SEQL; ++l) {
        const int row = b * SEQL + l;
        if (l + 2 < SEQL) {   // prefetch next activation rows (global_prefetch_b8)
            __builtin_prefetch(xpath + (size_t)(row + 2) * DINNER + e, 0, 0);
            __builtin_prefetch(zbuf  + (size_t)(row + 2) * DINNER + e, 0, 0);
        }
        // block-uniform addresses -> scalar loads
        const float dti = dt_in[row];
        const float lam = lam_arr[row];
        const float dt  = softplusf(dti * wdt + bdt);
        const float xp  = xpath[(size_t)row * DINNER + e];
        const float zv  = zbuf [(size_t)row * DINNER + e];
        const float* brp = Br + (size_t)row * DSTATE;
        const float* crp = Cr + (size_t)row * DSTATE;

        const float c1 = (l == 0) ? dt  : lam * dt;
        const float c0 = (l == 0) ? 0.f : (1.f - lam) * dt;

        float y = 0.f;
#pragma unroll
        for (int s = 0; s < DSTATE; ++s) {
            float dA = expf(A[s] * dt);
            float bx = brp[s] * xp;
            float u  = c0 * dA * bxp[s] + c1 * bx;
            h[s]   = dA * h[s] + u;
            y     += h[s] * crp[s];
            bxp[s] = bx;
        }
        float yo = (y + xp * Dv) * (zv * sigmoidf(zv));
        y_gated[(size_t)row * DINNER + e] = f32_to_bf16(yo);
    }
}

// ----------------------------------------------------------------------------
extern "C" void kernel_launch(void* const* d_in, const int* in_sizes, int n_in,
                              void* d_out, int out_size, void* d_ws, size_t ws_size,
                              hipStream_t stream) {
    const float* x          = (const float*)d_in[0];
    const float* W_in       = (const float*)d_in[1];
    const float* W_x        = (const float*)d_in[2];
    const float* W_dt       = (const float*)d_in[3];
    const float* b_dt       = (const float*)d_in[4];
    const float* A_log      = (const float*)d_in[5];
    const float* B_bias     = (const float*)d_in[6];
    const float* C_bias     = (const float*)d_in[7];
    const float* B_norm_w   = (const float*)d_in[8];
    const float* C_norm_w   = (const float*)d_in[9];
    const float* rope_freqs = (const float*)d_in[10];
    const float* Dvec       = (const float*)d_in[11];
    const float* W_out      = (const float*)d_in[12];
    float* out = (float*)d_out;

    char* ws = (char*)d_ws;
    size_t off = 0;
    auto carve = [&](size_t bytes) -> char* {
        char* p = ws + off;
        off = (off + bytes + 255) & ~(size_t)255;
        return p;
    };
    unsigned short* x_bf    = (unsigned short*)carve((size_t)NROWS * DMODEL * 2);
    unsigned short* Win_bf  = (unsigned short*)carve((size_t)(2 * DINNER) * DMODEL * 2);
    unsigned short* Wout_bf = (unsigned short*)carve((size_t)DMODEL * DINNER * 2);
    float* xpath   = (float*)carve((size_t)NROWS * DINNER * 4);
    float* zbuf    = (float*)carve((size_t)NROWS * DINNER * 4);
    float* Bn      = (float*)carve((size_t)NROWS * DSTATE * 4);
    float* Cn      = (float*)carve((size_t)NROWS * DSTATE * 4);
    float* Brw     = (float*)carve((size_t)NROWS * DSTATE * 4);
    float* Crw     = (float*)carve((size_t)NROWS * DSTATE * 4);
    float* dt_in   = (float*)carve((size_t)NROWS * 4);
    float* lam     = (float*)carve((size_t)NROWS * 4);
    float* mean_dt = (float*)carve((size_t)NROWS * 4);
    float* cum_dt  = (float*)carve((size_t)NROWS * 4);
    unsigned short* y_gated = (unsigned short*)carve((size_t)NROWS * DINNER * 2);

    // 1) bf16 conversions
    {
        int n1 = NROWS * DMODEL;                    // x
        convert_bf16_kernel<<<(n1 + 255) / 256, 256, 0, stream>>>(x, x_bf, n1);
        int n2 = 2 * DINNER * DMODEL;               // W_in
        convert_bf16_kernel<<<(n2 + 255) / 256, 256, 0, stream>>>(W_in, Win_bf, n2);
        int n3 = DMODEL * DINNER;                   // W_out
        convert_bf16_kernel<<<(n3 + 255) / 256, 256, 0, stream>>>(W_out, Wout_bf, n3);
    }

    // 2) GEMM1: xz = x · W_in^T  (silu -> xpath, raw -> z)
    wmma_gemm_kernel<0><<<dim3(NROWS / 128, (2 * DINNER) / 64), 256, 0, stream>>>(
        x_bf, Win_bf, xpath, zbuf, NROWS, 2 * DINNER, DMODEL);

    // 3) per-row ssm projection + rmsnorm + dt/lam/mean_dt
    ssm_row_kernel<<<NROWS, 256, 0, stream>>>(
        xpath, W_x, W_dt, b_dt, B_bias, C_bias, B_norm_w, C_norm_w,
        Bn, Cn, dt_in, lam, mean_dt);

    // 4) cumsum of mean dt along sequence (per batch)
    cumsum_kernel<<<1, 32, 0, stream>>>(mean_dt, cum_dt);

    // 5) RoPE on Bn/Cn
    rope_kernel<<<(NROWS + 255) / 256, 256, 0, stream>>>(
        Bn, Cn, cum_dt, rope_freqs, Brw, Crw);

    // 6) SSM scan (fused skip + gate), bf16 output
    scan_kernel<<<16, 256, 0, stream>>>(
        xpath, zbuf, Brw, Crw, dt_in, lam, W_dt, b_dt, A_log, Dvec, y_gated);

    // 7) GEMM2: out = y_gated · W_out^T
    wmma_gemm_kernel<1><<<dim3(NROWS / 128, DMODEL / 64), 256, 0, stream>>>(
        y_gated, Wout_bf, out, nullptr, NROWS, DMODEL, DINNER);
}
